// Attention_noZeromap_10084583211377
// MI455X (gfx1250) — matrix operand order repaired and checked by reading, not compiled, not run on hardware
//
#include <hip/hip_runtime.h>
#include <hip/hip_bf16.h>

// ---------------------------------------------------------------------------
// Problem constants (B,C,H,W) = (8,128,128,128)
// ---------------------------------------------------------------------------
#define NB   8
#define NC   128
#define NH   128
#define NW   128
#define HW   16384          // NH*NW
#define NPIX 131072         // NB*HW
#define O3   384            // 3*NC

typedef unsigned short u16;
typedef __attribute__((ext_vector_type(16))) __bf16        v16bf;
typedef __attribute__((ext_vector_type(8)))  float         v8f;
typedef __attribute__((ext_vector_type(4)))  unsigned int  v4u;
typedef __attribute__((ext_vector_type(4)))  unsigned int  u32x4;
typedef __attribute__((ext_vector_type(8)))  int           i32x8;
typedef __attribute__((ext_vector_type(4)))  int           i32x4;

union BF16Frag { v16bf v; u16 u[16]; v4u q[2]; };

#if defined(__has_builtin)
#if __has_builtin(__builtin_amdgcn_tensor_load_to_lds)
#define HAVE_TDM 1
#endif
#endif

__device__ __forceinline__ u16 f2bf(float f) {
  union { float f; unsigned int i; } c; c.f = f;
  unsigned int r = c.i + 0x7FFFu + ((c.i >> 16) & 1u);   // round-to-nearest-even
  return (u16)(r >> 16);
}

__device__ __forceinline__ unsigned lds_off(const void* p) {
  return (unsigned)(unsigned long long)p;                // low 32 bits = LDS byte offset
}

#ifdef HAVE_TDM
// 2D tile load: `rows` rows of 128 u16 elements from a (tensor_rows x 128) u16
// tensor (row stride 128) into LDS at byte offset `loff`.  D# per ISA §8.3/8.4.
// 6-arg builtin (clang-23 / therock-10.0): (g0, g1, g2, g3, g4, cpol).
__device__ __forceinline__ void tdm_load_tile(unsigned loff, const u16* gptr,
                                              unsigned rows, unsigned tensor_rows) {
  unsigned long long ga = (unsigned long long)gptr;
  u32x4 g0;
  g0[0] = 1u;                                            // count=1, user mode
  g0[1] = loff;                                          // lds_addr
  g0[2] = (unsigned)(ga & 0xFFFFFFFFu);                  // global_addr[31:0]
  g0[3] = (unsigned)((ga >> 32) & 0x01FFFFFFu) | (2u << 30);  // addr[56:32] | type=2
  i32x8 g1;
  g1[0] = (int)(1u << 16);                               // data_size = 1 (2 bytes)
  g1[1] = (int)(128u << 16);                             // tensor_dim0[15:0] = 128
  g1[2] = (int)((tensor_rows & 0xFFFFu) << 16);          // tensor_dim1 lo16
  g1[3] = (int)((tensor_rows >> 16) & 0xFFFFu) | (int)(128u << 16); // dim1 hi | tile_dim0=128
  g1[4] = (int)rows;                                     // tile_dim1
  g1[5] = 128;                                           // tensor_dim0_stride lo32
  g1[6] = 0;
  g1[7] = 0;
  i32x4 z4 = {};
  i32x8 z8 = {};
  __builtin_amdgcn_tensor_load_to_lds(g0, g1, z4, z4, z8, 0);
}
#endif

__device__ __forceinline__ float blk_sum128(float v, float* red, int tid) {
  red[tid] = v; __syncthreads();
  #pragma unroll
  for (int s = 64; s > 0; s >>= 1) { if (tid < s) red[tid] += red[tid + s]; __syncthreads(); }
  float r = red[0]; __syncthreads(); return r;
}
__device__ __forceinline__ float blk_max128(float v, float* red, int tid) {
  red[tid] = v; __syncthreads();
  #pragma unroll
  for (int s = 64; s > 0; s >>= 1) { if (tid < s) red[tid] = fmaxf(red[tid], red[tid + s]); __syncthreads(); }
  float r = red[0]; __syncthreads(); return r;
}

// ---------------------------------------------------------------------------
// fp32 -> bf16 (weights)
// ---------------------------------------------------------------------------
__global__ void k_f32_to_bf16(const float* __restrict__ in, u16* __restrict__ out, int n) {
  int i = blockIdx.x * blockDim.x + threadIdx.x;
  if (i < n) out[i] = f2bf(in[i]);
}

// ---------------------------------------------------------------------------
// LayerNorm over C per pixel; writes pixel-major bf16  xnT[pid*128 + c]
// ---------------------------------------------------------------------------
__global__ void k_ln(const float* __restrict__ x, const float* __restrict__ lnw,
                     const float* __restrict__ lnb, u16* __restrict__ xnT) {
  int pid = blockIdx.x * blockDim.x + threadIdx.x;          // 0..NPIX-1
  int b = pid >> 14; int p = pid & (HW - 1);
  const float* xb = x + (size_t)b * NC * HW + p;
  float s = 0.f, ss = 0.f;
  #pragma unroll 4
  for (int c = 0; c < NC; ++c) { float v = xb[(size_t)c * HW]; s += v; ss += v * v; }
  float mu = s * (1.f / NC);
  float var = ss * (1.f / NC) - mu * mu;
  float inv = rsqrtf(var + 1e-5f);
  u16* o = xnT + (size_t)pid * NC;
  #pragma unroll 4
  for (int c = 0; c < NC; ++c) {
    float v = (xb[(size_t)c * HW] - mu) * inv * lnw[c] + lnb[c];
    o[c] = f2bf(v);
  }
}

// ---------------------------------------------------------------------------
// 1x1-conv GEMM, TDM double-staged:
//   out[b,o,p] = sum_c Wbf[o,c] * XT[b*HW+p, c] + bias[o] (+res)
// Block = 4 waves. TDM pulls a 64x128 weight tile and a 64x128 pixel tile
// into LDS; each wave owns one 16-row M-tile x four 16-col N-tiles
// (A fragment reused across 4 WMMAs per K-step; 16 WMMAs per wave).
// ---------------------------------------------------------------------------
__global__ void k_gemm_tdm(const u16* __restrict__ Wbf, const u16* __restrict__ XT,
                           const float* __restrict__ bias, const float* __restrict__ residual,
                           float* __restrict__ out, int O, int mgroups) {
  __shared__ u16 lA[64 * 128];
  __shared__ u16 lB[64 * 128];
  int mg = blockIdx.x % mgroups;
  int ng = blockIdx.x / mgroups;
  int m0 = mg << 6;            // 64 output rows per block
  int P0 = ng << 6;            // 64 pixels per block

#ifdef HAVE_TDM
  if (threadIdx.x < 32) {
    tdm_load_tile(lds_off(lA), Wbf + (size_t)m0 * NC, 64, (unsigned)O);
    tdm_load_tile(lds_off(lB), XT + (size_t)P0 * NC, 64, (unsigned)NPIX);
    __builtin_amdgcn_s_wait_tensorcnt(0);
  }
#else
  {  // cooperative fallback: 128 threads copy 2 x 16KB
    const v4u* gA = (const v4u*)(Wbf + (size_t)m0 * NC);
    const v4u* gB = (const v4u*)(XT + (size_t)P0 * NC);
    v4u* sA = (v4u*)lA; v4u* sB = (v4u*)lB;
    #pragma unroll
    for (int i = 0; i < 8; ++i) {
      sA[threadIdx.x + 128 * i] = gA[threadIdx.x + 128 * i];
      sB[threadIdx.x + 128 * i] = gB[threadIdx.x + 128 * i];
    }
  }
#endif
  __syncthreads();

  int wave = threadIdx.x >> 5, lane = threadIdx.x & 31;
  int lr = lane & 15, lh = lane >> 4;
  const u16* As = &lA[(wave * 16 + lr) * NC];
  v8f acc[4] = {{}, {}, {}, {}};
  #pragma unroll
  for (int kk = 0; kk < 4; ++kk) {
    int k0 = kk << 5;
    BF16Frag a;
    int ka = k0 + lh * 8;
    a.q[0] = *(const v4u*)(As + ka);
    a.q[1] = *(const v4u*)(As + ka + 16);
    int kb = k0 + lh * 8;
    #pragma unroll
    for (int nt = 0; nt < 4; ++nt) {
      BF16Frag bq;
      const u16* Bs = &lB[(nt * 16 + lr) * NC];
      bq.q[0] = *(const v4u*)(Bs + kb);
      bq.q[1] = *(const v4u*)(Bs + kb + 16);
      acc[nt] = __builtin_amdgcn_wmma_f32_16x16x32_bf16(false, a.v, false, bq.v,
                                                        (short)0, acc[nt], false, false);
    }
  }
  int o0 = m0 + wave * 16;
  #pragma unroll
  for (int nt = 0; nt < 4; ++nt) {
    int Pa = P0 + nt * 16 + lr;
    int bb = Pa >> 14; int p = Pa & (HW - 1);
    #pragma unroll
    for (int j = 0; j < 8; ++j) {
      int o = o0 + lh * 8 + j;
      size_t oidx = ((size_t)bb * O + o) * HW + p;
      float val = acc[nt][j] + bias[o];
      if (residual) val += residual[oidx];
      out[oidx] = val;
    }
  }
}

// ---------------------------------------------------------------------------
// Depthwise 3x3, zero pad; fused bf16 side-output of the V channels
// vmode: 1 -> vbf[b,cv,h,w]    (pass 1)
//        2 -> vbf[b,cv,w,h]    (pass 2, transposed for attn2-apply B side)
// ---------------------------------------------------------------------------
__global__ void k_dwc(const float* __restrict__ in, const float* __restrict__ w9,
                      const float* __restrict__ bias, float* __restrict__ out,
                      u16* __restrict__ vbf, int vmode) {
  int idx = blockIdx.x;
  int h  = idx & (NH - 1);
  int ch = (idx >> 7) % O3;
  int b  = idx / (O3 * NH);
  int w  = threadIdx.x;
  const float* base = in + ((size_t)b * O3 + ch) * HW;
  const float* wk = w9 + ch * 9;
  float s = bias[ch];
  #pragma unroll
  for (int dy = -1; dy <= 1; ++dy) {
    int hh = h + dy; if (hh < 0 || hh >= NH) continue;
    #pragma unroll
    for (int dx = -1; dx <= 1; ++dx) {
      int ww = w + dx; if (ww < 0 || ww >= NW) continue;
      s += base[hh * NW + ww] * wk[(dy + 1) * 3 + (dx + 1)];
    }
  }
  out[((size_t)b * O3 + ch) * HW + h * NW + w] = s;
  if (ch >= 2 * NC) {
    int cv = ch - 2 * NC;
    size_t vb = ((size_t)b * NC + cv) * HW;
    if (vmode == 1) vbf[vb + h * NW + w] = f2bf(s);
    else            vbf[vb + w * NH + h] = f2bf(s);
  }
}

// ---------------------------------------------------------------------------
// Pass-1 reductions:  q1 = l2n(mean_h q),  k1 = l2n(max_h k)    (norm over W)
// ---------------------------------------------------------------------------
__global__ void k_reduce1(const float* __restrict__ qkvd, float* __restrict__ q1n,
                          float* __restrict__ k1n) {
  __shared__ float red[128];
  int blk = blockIdx.x; int b = blk >> 7, c = blk & 127, w = threadIdx.x;
  const float* qb = qkvd + ((size_t)b * O3 + c) * HW;
  const float* kb = qkvd + ((size_t)b * O3 + NC + c) * HW;
  float qs = 0.f, km = -INFINITY;
  for (int h = 0; h < NH; ++h) { qs += qb[h * NW + w]; km = fmaxf(km, kb[h * NW + w]); }
  float qm = qs * (1.f / NH);
  float qn2 = blk_sum128(qm * qm, red, w);
  float kn2 = blk_sum128(km * km, red, w);
  q1n[((size_t)b * NC + c) * NW + w] = qm / fmaxf(sqrtf(qn2), 1e-12f);
  k1n[((size_t)b * NC + c) * NW + w] = km / fmaxf(sqrtf(kn2), 1e-12f);
}

// attn1[b,w,v] = softmax_v( temp1 * sum_c q1n[b,c,w] k1n[b,c,v] )
__global__ void k_attn1(const float* __restrict__ q1n, const float* __restrict__ k1n,
                        const float* __restrict__ temp, float* __restrict__ attn) {
  __shared__ float red[128];
  int blk = blockIdx.x; int b = blk >> 7, w = blk & 127, v = threadIdx.x;
  float t = temp[0], s = 0.f;
  const float* qb = q1n + (size_t)b * NC * NW + w;
  const float* kb = k1n + (size_t)b * NC * NW + v;
  for (int c = 0; c < NC; ++c) s += qb[(size_t)c * NW] * kb[(size_t)c * NW];
  s *= t;
  float m = blk_max128(s, red, v);
  float e = __expf(s - m);
  float sum = blk_sum128(e, red, v);
  attn[((size_t)b * NW + w) * NW + v] = e / sum;
}

// Pass-2 reductions: q2/k2 = l2n(mean_c .) over W ; block per (b,h)
__global__ void k_reduce2(const float* __restrict__ qkvd, float* __restrict__ q2n,
                          float* __restrict__ k2n) {
  __shared__ float red[128];
  int blk = blockIdx.x; int b = blk >> 7, h = blk & 127, w = threadIdx.x;
  const float* base = qkvd + (size_t)b * O3 * HW + h * NW + w;
  float qs = 0.f, ks = 0.f;
  for (int c = 0; c < NC; ++c) {
    qs += base[(size_t)c * HW];
    ks += base[(size_t)(NC + c) * HW];
  }
  float qm = qs * (1.f / NC), km = ks * (1.f / NC);
  float qn2 = blk_sum128(qm * qm, red, w);
  float kn2 = blk_sum128(km * km, red, w);
  q2n[((size_t)b * NH + h) * NW + w] = qm / fmaxf(sqrtf(qn2), 1e-12f);
  k2n[((size_t)b * NH + h) * NW + w] = km / fmaxf(sqrtf(kn2), 1e-12f);
}

// attn2[b,h,g] = softmax_g( temp2 * sum_w q2n[b,h,w] k2n[b,g,w] )
__global__ void k_attn2(const float* __restrict__ q2n, const float* __restrict__ k2n,
                        const float* __restrict__ temp, float* __restrict__ attn) {
  __shared__ float red[128];
  int blk = blockIdx.x; int b = blk >> 7, h = blk & 127, g = threadIdx.x;
  float t = temp[0], s = 0.f;
  const float* qb = q2n + ((size_t)b * NH + h) * NW;
  const float* kb = k2n + ((size_t)b * NH + g) * NW;
  for (int w = 0; w < NW; ++w) s += qb[w] * kb[w];
  s *= t;
  float m = blk_max128(s, red, g);
  float e = __expf(s - m);
  float sum = blk_sum128(e, red, g);
  attn[((size_t)b * NH + h) * NH + g] = e / sum;
}

// transpose 128x128 per batch + cvt to bf16:  out[b,i,j] = bf16(in[b,j,i])
__global__ void k_transpose_cvt(const float* __restrict__ in, u16* __restrict__ out) {
  int blk = blockIdx.x; int b = blk >> 7, i = blk & 127, j = threadIdx.x;
  out[((size_t)b * 128 + i) * 128 + j] = f2bf(in[((size_t)b * 128 + j) * 128 + i]);
}

// ---------------------------------------------------------------------------
// attn1 apply: TT[b, h*128+v, c] = bf16( sum_w vbf[b,c,h,w] * aT[b,v,w] )
// Wave: 4 M-tiles (h0..h0+63, fixed c) x one N-tile (v); B fragment reused.
// ---------------------------------------------------------------------------
__global__ void k_attn1_apply(const u16* __restrict__ vbf, const u16* __restrict__ aT,
                              u16* __restrict__ TT) {
  int wave = threadIdx.x >> 5, lane = threadIdx.x & 31;
  int gid = blockIdx.x * (blockDim.x >> 5) + wave;          // 0..16383
  int b = gid >> 11; int t = gid & 2047;
  int vt = t & 7, hq = (t >> 3) & 1, c = t >> 4;
  int h0 = hq << 6, v0 = vt << 4;
  int lr = lane & 15, lh = lane >> 4;

  const u16* Abase = vbf + (((size_t)b * NC + c) * NH + h0 + lr) * NW;
  const u16* Brow  = aT + ((size_t)b * NW + v0 + lr) * NW;
  v8f acc[4] = {{}, {}, {}, {}};
  #pragma unroll
  for (int kk = 0; kk < 4; ++kk) {
    int k0 = kk << 5;
    int ka = k0 + lh * 8;
    int kb = k0 + lh * 8;
    BF16Frag bq;
    bq.q[0] = *(const v4u*)(Brow + kb);
    bq.q[1] = *(const v4u*)(Brow + kb + 16);
    #pragma unroll
    for (int mi = 0; mi < 4; ++mi) {
      BF16Frag a;
      const u16* Arow = Abase + (size_t)(mi * 16) * NW;
      a.q[0] = *(const v4u*)(Arow + ka);
      a.q[1] = *(const v4u*)(Arow + ka + 16);
      acc[mi] = __builtin_amdgcn_wmma_f32_16x16x32_bf16(false, a.v, false, bq.v,
                                                        (short)0, acc[mi], false, false);
    }
  }
  int vv = v0 + lr;
  #pragma unroll
  for (int mi = 0; mi < 4; ++mi) {
    #pragma unroll
    for (int j = 0; j < 8; ++j) {
      int hh = h0 + mi * 16 + lh * 8 + j;
      TT[((size_t)b * HW + hh * NW + vv) * NC + c] = f2bf(acc[mi][j]);
    }
  }
}

// ---------------------------------------------------------------------------
// attn2 apply: TT[b, g*128+w, c] = bf16( sum_h aT2[b,g,h] * v2T[b,c,w,h] )
// Wave: one M-tile (g) x 4 N-tiles (w0..w0+63, fixed c); A fragment reused.
// v2T transposed layout -> B fragments are contiguous 128-bit loads.
// ---------------------------------------------------------------------------
__global__ void k_attn2_apply(const u16* __restrict__ v2T, const u16* __restrict__ aT2,
                              u16* __restrict__ TT) {
  int wave = threadIdx.x >> 5, lane = threadIdx.x & 31;
  int gid = blockIdx.x * (blockDim.x >> 5) + wave;          // 0..16383
  int bc = gid >> 4; int b = bc >> 7, c = bc & 127;
  int t = gid & 15;  int g0 = (t & 7) << 4, w0 = (t >> 3) << 6;
  int lr = lane & 15, lh = lane >> 4;

  const u16* Arow  = aT2 + ((size_t)b * NH + g0 + lr) * NH;           // row g, contiguous h
  const u16* Bbase = v2T + ((size_t)b * NC + c) * HW + (size_t)(w0 + lr) * NH;
  v8f acc[4] = {{}, {}, {}, {}};
  #pragma unroll
  for (int kk = 0; kk < 4; ++kk) {
    int k0 = kk << 5;
    int ka = k0 + lh * 8;
    int kb = k0 + lh * 8;
    BF16Frag a;
    a.q[0] = *(const v4u*)(Arow + ka);
    a.q[1] = *(const v4u*)(Arow + ka + 16);
    #pragma unroll
    for (int nt = 0; nt < 4; ++nt) {
      BF16Frag bq;
      const u16* Brow = Bbase + (size_t)(nt * 16) * NH;
      bq.q[0] = *(const v4u*)(Brow + kb);
      bq.q[1] = *(const v4u*)(Brow + kb + 16);
      acc[nt] = __builtin_amdgcn_wmma_f32_16x16x32_bf16(false, a.v, false, bq.v,
                                                        (short)0, acc[nt], false, false);
    }
  }
  #pragma unroll
  for (int nt = 0; nt < 4; ++nt) {
    int ww = w0 + nt * 16 + lr;
    #pragma unroll
    for (int j = 0; j < 8; ++j) {
      int gg = g0 + lh * 8 + j;
      TT[((size_t)b * HW + gg * NW + ww) * NC + c] = f2bf(acc[nt][j]);
    }
  }
}

// ---------------------------------------------------------------------------
// Workspace layout (bytes, 256B aligned)
// ---------------------------------------------------------------------------
static constexpr size_t OFF_XNT   = 0;                      // bf16 NPIX*NC — xnT / vbf / v2T
static constexpr size_t OFF_QKV   = 33554432;               // f32 NB*O3*HW — aliased as TT (bf16)
static constexpr size_t OFF_QKVD  = 234881024;              // f32 NB*O3*HW
static constexpr size_t OFF_OUT1  = 436207616;              // f32 NB*NC*HW
static constexpr size_t OFF_WQKV  = 503316480;              // bf16 384*128
static constexpr size_t OFF_WPROJ = 503414784;              // bf16 128*128
static constexpr size_t OFF_Q1N   = 503447552;              // f32 NB*128*128
static constexpr size_t OFF_K1N   = 503971840;
static constexpr size_t OFF_ATT   = 504496128;              // f32 NB*128*128
static constexpr size_t OFF_ATTT  = 505020416;              // bf16 NB*128*128

extern "C" void kernel_launch(void* const* d_in, const int* in_sizes, int n_in,
                              void* d_out, int out_size, void* d_ws, size_t ws_size,
                              hipStream_t stream) {
  (void)in_sizes; (void)n_in; (void)out_size; (void)ws_size;
  const float* x      = (const float*)d_in[0];
  const float* ln_w   = (const float*)d_in[1];
  const float* ln_b   = (const float*)d_in[2];
  const float* qkv_w  = (const float*)d_in[3];
  const float* qkv_b  = (const float*)d_in[4];
  const float* dw_w   = (const float*)d_in[5];
  const float* dw_b   = (const float*)d_in[6];
  const float* proj_w = (const float*)d_in[7];
  const float* proj_b = (const float*)d_in[8];
  const float* temp1  = (const float*)d_in[9];
  const float* temp2  = (const float*)d_in[10];

  char* ws = (char*)d_ws;
  u16*   xnT   = (u16*)  (ws + OFF_XNT);     // also vbf / v2T
  float* qkv   = (float*)(ws + OFF_QKV);
  u16*   TT    = (u16*)  (ws + OFF_QKV);     // alias (disjoint lifetime)
  float* qkvd  = (float*)(ws + OFF_QKVD);
  float* out1  = (float*)(ws + OFF_OUT1);
  u16*   wqkv  = (u16*)  (ws + OFF_WQKV);
  u16*   wproj = (u16*)  (ws + OFF_WPROJ);
  float* q1n   = (float*)(ws + OFF_Q1N);
  float* k1n   = (float*)(ws + OFF_K1N);
  float* attn  = (float*)(ws + OFF_ATT);
  u16*   attnT = (u16*)  (ws + OFF_ATTT);

  // weight conversion
  k_f32_to_bf16<<<192, 256, 0, stream>>>(qkv_w, wqkv, O3 * NC);
  k_f32_to_bf16<<<64,  256, 0, stream>>>(proj_w, wproj, NC * NC);

  // ---- pass 1 ----
  k_ln<<<NPIX / 128, 128, 0, stream>>>(x, ln_w, ln_b, xnT);
  k_gemm_tdm<<<6 * 2048, 128, 0, stream>>>(wqkv, xnT, qkv_b, nullptr, qkv, O3, 6);
  k_dwc<<<NB * O3 * NH, 128, 0, stream>>>(qkv, dw_w, dw_b, qkvd, xnT, 1);   // + v1 -> bf16
  k_reduce1<<<NB * NC, 128, 0, stream>>>(qkvd, q1n, k1n);
  k_attn1<<<NB * NW, 128, 0, stream>>>(q1n, k1n, temp1, attn);
  k_transpose_cvt<<<NB * 128, 128, 0, stream>>>(attn, attnT);
  k_attn1_apply<<<4096, 128, 0, stream>>>(xnT, attnT, TT);
  k_gemm_tdm<<<2 * 2048, 128, 0, stream>>>(wproj, TT, proj_b, x, out1, NC, 2);

  // ---- pass 2 ----
  k_ln<<<NPIX / 128, 128, 0, stream>>>(out1, ln_w, ln_b, xnT);
  k_gemm_tdm<<<6 * 2048, 128, 0, stream>>>(wqkv, xnT, qkv_b, nullptr, qkv, O3, 6);
  k_dwc<<<NB * O3 * NH, 128, 0, stream>>>(qkv, dw_w, dw_b, qkvd, xnT, 2);   // + v2T (transposed)
  k_reduce2<<<NB * NH, 128, 0, stream>>>(qkvd, q1n, k1n);                   // q2n/k2n
  k_attn2<<<NB * NH, 128, 0, stream>>>(q1n, k1n, temp2, attn);
  k_transpose_cvt<<<NB * 128, 128, 0, stream>>>(attn, attnT);
  k_attn2_apply<<<4096, 128, 0, stream>>>(xnT, attnT, TT);
  k_gemm_tdm<<<2 * 2048, 128, 0, stream>>>(wproj, TT, proj_b, out1, (float*)d_out, NC, 2);
}
